// ACMultiLayerdLSTM_73632919323053
// MI455X (gfx1250) — matrix operand order repaired and checked by reading, not compiled, not run on hardware
//
#include <hip/hip_runtime.h>
#include <hip/hip_bf16.h>

// ACMultiLayerdLSTM for MI455X (gfx1250): bf16 WMMA GEMMs, fp32 cell state.
// B=64, T=512, F=256, U=1024, L=3 layers; gates (i,j,f,o), forget_bias=1.0.

typedef __attribute__((ext_vector_type(16))) __bf16 v16bf;
typedef __attribute__((ext_vector_type(8)))  __bf16 v8bf;
typedef __attribute__((ext_vector_type(8)))  float  v8f;

#define B_ 64
#define T_ 512
#define F_ 256
#define U_ 1024
#define HS_ (B_ * U_)   // one layer's h/c matrix: 65536 elements

__device__ __forceinline__ float sigm(float x) { return 1.0f / (1.0f + __expf(-x)); }

__device__ __forceinline__ v8bf load8(const __bf16* p) { return *(const v8bf*)p; }

__device__ __forceinline__ v16bf combine16(v8bf lo, v8bf hi) {
    v16bf r;
#pragma unroll
    for (int i = 0; i < 8; ++i) { r[i] = lo[i]; r[i + 8] = hi[i]; }
    return r;
}

// A-matrix element pointer: rows are [x-part (width Kx) | h-part (width U_)]
__device__ __forceinline__ const __bf16* seg_ptr(const __bf16* __restrict__ xp, int Kx,
                                                 const __bf16* __restrict__ hp,
                                                 int brow, int k) {
    return (k < Kx) ? (xp + (size_t)brow * Kx + k)
                    : (hp + (size_t)brow * U_ + (k - Kx));
}

// ---------------------------------------------------------------------------
// One LSTM layer for one timestep. 256 waves: wave = (mt in 0..3, ut in 0..63).
// Each wave accumulates four 16x16 gate tiles (i,j,f,o) over K with
// v_wmma_f32_16x16x32_bf16 and does the cell update in registers.
// Wt is the gate weight matrix transposed to [4096][Ktot] bf16.
// ---------------------------------------------------------------------------
__global__ __launch_bounds__(256)
void lstm_layer_kernel(const __bf16* __restrict__ xpart, int Kx,
                       const __bf16* __restrict__ hprev,
                       const __bf16* __restrict__ Wt, int Ktot,
                       const float* __restrict__ bias,
                       float* __restrict__ c,
                       __bf16* __restrict__ hout) {
    const int lane = threadIdx.x & 31;
    const int wid  = (blockIdx.x * blockDim.x + threadIdx.x) >> 5;
    const int mt   = wid & 3;     // batch tile (B=64 -> 4 tiles)
    const int ut   = wid >> 2;    // unit tile  (U=1024 -> 64 tiles)
    const int hlf  = lane >> 4;   // half-wave selector per ISA fragment layout
    const int l16  = lane & 15;

    const int brow  = mt * 16 + l16;   // A-matrix row held by this lane
    const int ncol0 = ut * 16 + l16;   // unit column within each gate block

    v8f acc[4];
#pragma unroll
    for (int g = 0; g < 4; ++g)
#pragma unroll
        for (int r = 0; r < 8; ++r) acc[g][r] = 0.0f;

    const int nkt = Ktot >> 5;   // K tiles of 32
    for (int kt = 0; kt < nkt; ++kt) {
        const int k0 = kt << 5;
        // A fragment (16x32 bf16): lane<16 -> K {k0..k0+7, k0+16..k0+23}; lane>=16 -> +8
        const int ka = k0 + hlf * 8;
        const v16bf afrag = combine16(load8(seg_ptr(xpart, Kx, hprev, brow, ka)),
                                      load8(seg_ptr(xpart, Kx, hprev, brow, ka + 16)));
        // B fragments (32x16 bf16): lane&15 = N column; K same as A fragment (lane>=16 -> +8).
        const int kb = k0 + hlf * 8;
#pragma unroll
        for (int g = 0; g < 4; ++g) {
            const __bf16* pb = Wt + (size_t)(g * U_ + ncol0) * Ktot + kb;
            const v16bf bfrag = combine16(load8(pb), load8(pb + 16));
            acc[g] = __builtin_amdgcn_wmma_f32_16x16x32_bf16(
                false, afrag, false, bfrag, (short)0, acc[g], false, false);
        }
    }

    const float bi = bias[0 * U_ + ncol0];
    const float bj = bias[1 * U_ + ncol0];
    const float bf = bias[2 * U_ + ncol0];
    const float bo = bias[3 * U_ + ncol0];

    // D layout: VGPR r, this lane -> row = mt*16 + r + 8*hlf, col = ncol0
#pragma unroll
    for (int r = 0; r < 8; ++r) {
        const int row = mt * 16 + r + 8 * hlf;
        const size_t idx = (size_t)row * U_ + ncol0;
        const float iv = acc[0][r] + bi;
        const float jv = acc[1][r] + bj;
        const float fv = acc[2][r] + bf;
        const float ov = acc[3][r] + bo;
        const float cold = c[idx];
        const float cnew = cold * sigm(fv + 1.0f) + sigm(iv) * tanhf(jv);
        const float hnew = sigm(ov) * tanhf(cnew);
        c[idx]    = cnew;
        hout[idx] = (__bf16)hnew;
    }
}

// ---------------------------------------------------------------------------
// Dense projection y = h2 @ Wd + bd : [64x1024]x[1024x256]. 64 waves.
// Wdt is Wd transposed to [256][1024] bf16. y written fp32.
// ---------------------------------------------------------------------------
__global__ __launch_bounds__(256)
void dense_kernel(const __bf16* __restrict__ h2,
                  const __bf16* __restrict__ Wdt,
                  const float* __restrict__ bd,
                  float* __restrict__ y) {
    const int lane = threadIdx.x & 31;
    const int wid  = (blockIdx.x * blockDim.x + threadIdx.x) >> 5;
    const int mt   = wid & 3;
    const int ft   = wid >> 2;   // 16 feature tiles
    const int hlf  = lane >> 4;
    const int l16  = lane & 15;
    const int brow = mt * 16 + l16;
    const int fcol = ft * 16 + l16;

    v8f acc;
#pragma unroll
    for (int r = 0; r < 8; ++r) acc[r] = 0.0f;

    for (int kt = 0; kt < (U_ >> 5); ++kt) {
        const int k0 = kt << 5;
        const __bf16* pa = h2 + (size_t)brow * U_ + k0 + hlf * 8;
        const v16bf afrag = combine16(load8(pa), load8(pa + 16));
        const __bf16* pb = Wdt + (size_t)fcol * U_ + k0 + hlf * 8;
        const v16bf bfrag = combine16(load8(pb), load8(pb + 16));
        acc = __builtin_amdgcn_wmma_f32_16x16x32_bf16(
            false, afrag, false, bfrag, (short)0, acc, false, false);
    }

    const float bv = bd[fcol];
#pragma unroll
    for (int r = 0; r < 8; ++r) {
        const int row = mt * 16 + r + 8 * hlf;
        y[(size_t)row * F_ + fcol] = acc[r] + bv;
    }
}

// ---------------------------------------------------------------------------
// LayerNorm + ReLU over F=256 per batch row; writes output slice and the emit
// buffer (fp32) used for auto-conditioning at the next step.
// ---------------------------------------------------------------------------
__global__ __launch_bounds__(256)
void ln_kernel(const float* __restrict__ y,
               const float* __restrict__ gamma,
               const float* __restrict__ beta,
               float* __restrict__ out,     // [B,T,F]
               float* __restrict__ emit,    // [B,F]
               int t) {
    __shared__ float sh[F_];
    const int b = blockIdx.x;
    const int f = threadIdx.x;
    const float v = y[b * F_ + f];
    sh[f] = v;
    __syncthreads();
    for (int s = 128; s > 0; s >>= 1) { if (f < s) sh[f] += sh[f + s]; __syncthreads(); }
    const float mu = sh[0] * (1.0f / F_);
    __syncthreads();
    const float d = v - mu;
    sh[f] = d * d;
    __syncthreads();
    for (int s = 128; s > 0; s >>= 1) { if (f < s) sh[f] += sh[f + s]; __syncthreads(); }
    const float var = sh[0] * (1.0f / F_);
    const float yn  = d * rsqrtf(var + 1e-12f) * gamma[f] + beta[f];
    const float e   = fmaxf(yn, 0.0f);
    out[(size_t)b * (T_ * F_) + (size_t)t * F_ + f] = e;
    emit[b * F_ + f] = e;
}

// x_t = (t==0 || cond[t]) ? input[:,t,:] : prev_emit ; cast to bf16
__global__ __launch_bounds__(256)
void select_x_kernel(const float* __restrict__ input,
                     const unsigned char* __restrict__ cond,
                     const float* __restrict__ emit,
                     __bf16* __restrict__ x, int t) {
    const int i = blockIdx.x * blockDim.x + threadIdx.x;   // 0..16383
    const int b = i >> 8, f = i & 255;
    const bool teach = (t == 0) || (cond[t] != 0);
    const float v = teach ? input[(size_t)b * (T_ * F_) + (size_t)t * F_ + f]
                          : emit[i];
    x[i] = (__bf16)v;
}

// Transpose-convert fp32 weight [K][N] -> bf16 [N][K] (coalesced source reads)
__global__ __launch_bounds__(256)
void convt_kernel(const float* __restrict__ src, __bf16* __restrict__ dst,
                  int K, int N) {
    const size_t i = (size_t)blockIdx.x * blockDim.x + threadIdx.x;
    if (i >= (size_t)K * N) return;
    const int n = (int)(i % N);
    const int k = (int)(i / N);
    dst[(size_t)n * K + k] = (__bf16)src[i];
}

__global__ __launch_bounds__(256)
void zero_state_kernel(float* __restrict__ c, __bf16* __restrict__ hA, int n) {
    const int i = blockIdx.x * blockDim.x + threadIdx.x;
    if (i < n) { c[i] = 0.0f; hA[i] = (__bf16)0.0f; }
}

extern "C" void kernel_launch(void* const* d_in, const int* in_sizes, int n_in,
                              void* d_out, int out_size, void* d_ws, size_t ws_size,
                              hipStream_t stream) {
    (void)in_sizes; (void)n_in; (void)out_size; (void)ws_size;
    const float*         input = (const float*)d_in[0];
    const unsigned char* cond  = (const unsigned char*)d_in[1];
    const float* W0 = (const float*)d_in[2];
    const float* b0 = (const float*)d_in[3];
    const float* W1 = (const float*)d_in[4];
    const float* b1 = (const float*)d_in[5];
    const float* W2 = (const float*)d_in[6];
    const float* b2 = (const float*)d_in[7];
    const float* Wd = (const float*)d_in[8];
    const float* bd = (const float*)d_in[9];
    const float* gamma = (const float*)d_in[10];
    const float* beta  = (const float*)d_in[11];
    float* out = (float*)d_out;

    // workspace carve-up (256B aligned)
    char* ws = (char*)d_ws;
    size_t off = 0;
    auto carve = [&](size_t bytes) -> char* {
        char* p = ws + off;
        off += (bytes + 255) & ~(size_t)255;
        return p;
    };
    __bf16* W0t  = (__bf16*)carve((size_t)4096 * 1280 * 2);
    __bf16* W1t  = (__bf16*)carve((size_t)4096 * 2048 * 2);
    __bf16* W2t  = (__bf16*)carve((size_t)4096 * 2048 * 2);
    __bf16* Wdt  = (__bf16*)carve((size_t)F_ * U_ * 2);
    __bf16* hA   = (__bf16*)carve((size_t)3 * HS_ * 2);
    __bf16* hB   = (__bf16*)carve((size_t)3 * HS_ * 2);
    float*  cbuf = (float*) carve((size_t)3 * HS_ * 4);
    __bf16* xb   = (__bf16*)carve((size_t)B_ * F_ * 2);
    float*  emit = (float*) carve((size_t)B_ * F_ * 4);
    float*  ybuf = (float*) carve((size_t)B_ * F_ * 4);

    // one-time (per launch) weight conversion + state zero-init
    convt_kernel<<<(1280u * 4096u + 255) / 256, 256, 0, stream>>>(W0, W0t, 1280, 4096);
    convt_kernel<<<(2048u * 4096u + 255) / 256, 256, 0, stream>>>(W1, W1t, 2048, 4096);
    convt_kernel<<<(2048u * 4096u + 255) / 256, 256, 0, stream>>>(W2, W2t, 2048, 4096);
    convt_kernel<<<(1024u * 256u  + 255) / 256, 256, 0, stream>>>(Wd, Wdt, 1024, 256);
    zero_state_kernel<<<(3 * HS_ + 255) / 256, 256, 0, stream>>>(cbuf, hA, 3 * HS_);

    for (int t = 0; t < T_; ++t) {
        __bf16* hprev = (t & 1) ? hB : hA;
        __bf16* hcur  = (t & 1) ? hA : hB;
        select_x_kernel<<<64, 256, 0, stream>>>(input, cond, emit, xb, t);
        lstm_layer_kernel<<<32, 256, 0, stream>>>(xb, F_, hprev + 0 * HS_,
                                                  W0t, F_ + U_, b0,
                                                  cbuf + 0 * HS_, hcur + 0 * HS_);
        lstm_layer_kernel<<<32, 256, 0, stream>>>(hcur + 0 * HS_, U_, hprev + 1 * HS_,
                                                  W1t, 2 * U_, b1,
                                                  cbuf + 1 * HS_, hcur + 1 * HS_);
        lstm_layer_kernel<<<32, 256, 0, stream>>>(hcur + 1 * HS_, U_, hprev + 2 * HS_,
                                                  W2t, 2 * U_, b2,
                                                  cbuf + 2 * HS_, hcur + 2 * HS_);
        dense_kernel<<<8, 256, 0, stream>>>(hcur + 2 * HS_, Wdt, bd, ybuf);
        ln_kernel<<<64, 256, 0, stream>>>(ybuf, gamma, beta, out, emit, t);
    }
}
